// CrossFormerBlock_86509231276254
// MI455X (gfx1250) — matrix operand order repaired and checked by reading, not compiled, not run on hardware
//
#include <hip/hip_runtime.h>

typedef __attribute__((ext_vector_type(16))) __bf16 v16bf;
typedef __attribute__((ext_vector_type(8)))  __bf16 v8bf;
typedef __attribute__((ext_vector_type(8)))  float  v8f;

#define WMMA_BF16(a, b, c) \
  __builtin_amdgcn_wmma_f32_16x16x32_bf16(false, (a), false, (b), (short)0, (c), false, false)

// ---------------------------------------------------------------------------
// A-fragment (16x32 bf16, M rows x K cols, row-major source, stride in elems)
// lanes 0-15: M=lane, K in {half*16..+15}  (half = lane>>4)
// ---------------------------------------------------------------------------
__device__ __forceinline__ v16bf load_a_frag(const __bf16* base, int stride,
                                             int m0, int k0, int lane) {
  const __bf16* p = base + (size_t)(m0 + (lane & 15)) * stride + k0 + ((lane >> 4) << 4);
  v8bf lo = *(const v8bf*)p;
  v8bf hi = *(const v8bf*)(p + 8);
  v16bf r;
#pragma unroll
  for (int i = 0; i < 8; ++i) { r[i] = lo[i]; r[i + 8] = hi[i]; }
  return r;
}

// ---------------------------------------------------------------------------
// B-fragment (32x16 bf16, K rows x N cols) gathered from a K-major operand:
// src[n][k] contiguous in k.  lanes 0-15: N=lane, K=0-15; lanes 16-31: K=16-31.
// ---------------------------------------------------------------------------
__device__ __forceinline__ v16bf load_b_frag(const __bf16* base, int stride,
                                             int n0, int k0, int lane) {
  const __bf16* p = base + (size_t)(n0 + (lane & 15)) * stride + k0 + ((lane >> 4) << 4);
  return *(const v16bf*)p;
}

// ---------------------------------------------------------------------------
// Prep 1: fp32 -> bf16 weight conversion
// ---------------------------------------------------------------------------
__global__ void conv_bf16_kernel(const float* __restrict__ w, __bf16* __restrict__ wb, int n) {
  int i = blockIdx.x * blockDim.x + threadIdx.x;
  if (i < n) wb[i] = (__bf16)w[i];
}

// ---------------------------------------------------------------------------
// Prep 2: DynamicPosBias MLP -> pos[961][4]
// ---------------------------------------------------------------------------
__device__ __forceinline__ void ln_relu8(const float* vin, float* t,
                                         const float* g, const float* b) {
  float m = 0.f;
#pragma unroll
  for (int p = 0; p < 8; ++p) m += vin[p];
  m *= 0.125f;
  float var = 0.f;
#pragma unroll
  for (int p = 0; p < 8; ++p) { float d = vin[p] - m; var += d * d; }
  var *= 0.125f;
  float inv = rsqrtf(var + 1e-5f);
#pragma unroll
  for (int p = 0; p < 8; ++p) t[p] = fmaxf((vin[p] - m) * inv * g[p] + b[p], 0.f);
}

__global__ void dpb_kernel(const float* __restrict__ pp_w, const float* __restrict__ pp_b,
                           const float* __restrict__ l1_g, const float* __restrict__ l1_b,
                           const float* __restrict__ f1_w, const float* __restrict__ f1_b,
                           const float* __restrict__ l2_g, const float* __restrict__ l2_b,
                           const float* __restrict__ f2_w, const float* __restrict__ f2_b,
                           const float* __restrict__ l3_g, const float* __restrict__ l3_b,
                           const float* __restrict__ f3_w, const float* __restrict__ f3_b,
                           float* __restrict__ pos) {
  int r = blockIdx.x * blockDim.x + threadIdx.x;
  if (r >= 961) return;
  float b0 = (float)(r / 31 - 15);
  float b1 = (float)(r % 31 - 15);
  float v0[8], v1[8], v2[8], t[8];
#pragma unroll
  for (int p = 0; p < 8; ++p) v0[p] = b0 * pp_w[p * 2 + 0] + b1 * pp_w[p * 2 + 1] + pp_b[p];
  ln_relu8(v0, t, l1_g, l1_b);
#pragma unroll
  for (int p = 0; p < 8; ++p) { float a = f1_b[p];
#pragma unroll
    for (int q = 0; q < 8; ++q) a += t[q] * f1_w[p * 8 + q];
    v1[p] = a; }
  ln_relu8(v1, t, l2_g, l2_b);
#pragma unroll
  for (int p = 0; p < 8; ++p) { float a = f2_b[p];
#pragma unroll
    for (int q = 0; q < 8; ++q) a += t[q] * f2_w[p * 8 + q];
    v2[p] = a; }
  ln_relu8(v2, t, l3_g, l3_b);
#pragma unroll
  for (int hh = 0; hh < 4; ++hh) { float a = f3_b[hh];
#pragma unroll
    for (int q = 0; q < 8; ++q) a += t[q] * f3_w[hh * 8 + q];
    pos[r * 4 + hh] = a; }
}

// ---------------------------------------------------------------------------
// Main: one block per window (16x16 tokens), 256 threads = 8 waves (wave32).
// ---------------------------------------------------------------------------
__global__ void __launch_bounds__(256) cf_attn_kernel(
    const float* __restrict__ x, const float* __restrict__ n1g, const float* __restrict__ n1b,
    const __bf16* __restrict__ wqkv, const float* __restrict__ qkv_b,
    const float* __restrict__ pos, float* __restrict__ out) {
  __shared__ __bf16 sX[256 * 128];   // LN'd window tile (bf16), 64 KB
  __shared__ __bf16 sQh[256 * 32];   // Q for current head (scaled), 16 KB
  __shared__ __bf16 sKh[256 * 32];   // K for current head, 16 KB
  __shared__ __bf16 sVTh[32 * 256];  // V transposed [dim][token], 16 KB
  __shared__ __bf16 sS[64 * 256];    // softmax row-block, 32 KB
  __shared__ float  sRed[256];       // softmax partial max/sum, 1 KB

  const int blk = blockIdx.x;
  const int b = blk >> 8;          // batch
  const int w = blk & 255;         // window id
  const int wi = w >> 4, wj = w & 15;
  const int t = threadIdx.x;
  const int lane = t & 31;
  const int wave = t >> 5;

  // ---- Stage 1: LayerNorm -> bf16 tile in LDS (one token per thread) ----
  {
    const int i = t >> 4, j = t & 15;
    const size_t rowoff =
        ((size_t)b * 65536 + (size_t)(wi * 16 + i) * 256 + (wj * 16 + j)) * 128;
    const float* row = x + rowoff;
    float s = 0.f, s2 = 0.f;
#pragma unroll
    for (int c = 0; c < 128; c += 4) {
      float4 f = *(const float4*)(row + c);
      s += f.x + f.y + f.z + f.w;
      s2 += f.x * f.x + f.y * f.y + f.z * f.z + f.w * f.w;
    }
    float mean = s * (1.f / 128.f);
    float var = s2 * (1.f / 128.f) - mean * mean;
    float inv = rsqrtf(var + 1e-5f);
#pragma unroll
    for (int c = 0; c < 128; c += 8) {
      float4 f0 = *(const float4*)(row + c);
      float4 f1 = *(const float4*)(row + c + 4);
      float fv[8] = {f0.x, f0.y, f0.z, f0.w, f1.x, f1.y, f1.z, f1.w};
      v8bf o;
#pragma unroll
      for (int u = 0; u < 8; ++u)
        o[u] = (__bf16)(((fv[u] - mean) * inv) * n1g[c + u] + n1b[c + u]);
      *(v8bf*)(&sX[t * 128 + c]) = o;
    }
  }
  __syncthreads();

  // ---- Per-head pipeline ----
  for (int head = 0; head < 4; ++head) {
    // Stage 2: QKV projection for this head's 96 channels.
    // tn unrolled -> store target (q/k/v) is compile-time; B-frags hoisted.
#pragma unroll
    for (int tn = 0; tn < 6; ++tn) {
      const int sec = tn >> 1;  // 0=q, 1=k, 2=v (constant after unroll)
      const int colbase = sec * 128 + head * 32 + (tn & 1) * 16;
      v16bf bfr[4];
#pragma unroll
      for (int kc = 0; kc < 4; ++kc)
        bfr[kc] = load_b_frag(wqkv, 128, colbase, kc * 32, lane);
      const float bias = qkv_b[colbase + (lane & 15)];
      const int lcol = (tn & 1) * 16 + (lane & 15);
      for (int tm = wave; tm < 16; tm += 8) {
        v8f acc = {};
#pragma unroll
        for (int kc = 0; kc < 4; ++kc) {
          v16bf a = load_a_frag(sX, 128, tm * 16, kc * 32, lane);
          acc = WMMA_BF16(a, bfr[kc], acc);
        }
#pragma unroll
        for (int r = 0; r < 8; ++r) {
          int mrow = tm * 16 + r + ((lane >> 4) << 3);
          float val = acc[r] + bias;
          if (sec == 0)      sQh[mrow * 32 + lcol] = (__bf16)(val * 0.17677669529663688f);
          else if (sec == 1) sKh[mrow * 32 + lcol] = (__bf16)val;
          else               sVTh[lcol * 256 + mrow] = (__bf16)val;
        }
      }
    }
    __syncthreads();

    // Stage 3: attention in 64-query-row blocks
    for (int rb = 0; rb < 4; ++rb) {
      int row0 = rb * 64;

      // S = Q K^T (+ bias): wave owns one Q row-tile, sweeps 8 key tiles.
      {
        const int tm = wave >> 1;
        const int tnbase = (wave & 1) * 8;
        v16bf a = load_a_frag(sQh, 32, row0 + tm * 16, 0, lane);
#pragma unroll
        for (int tj = 0; tj < 8; ++tj) {
          int tn = tnbase + tj;
          v16bf bb = load_b_frag(sKh, 32, tn * 16, 0, lane);
          v8f acc = {};
          acc = WMMA_BF16(a, bb, acc);
#pragma unroll
          for (int r = 0; r < 8; ++r) {
            int lr = tm * 16 + r + ((lane >> 4) << 3);
            int qt = row0 + lr;
            int kt = tn * 16 + (lane & 15);
            int di = (qt >> 4) - (kt >> 4) + 15;
            int dj = (qt & 15) - (kt & 15) + 15;
            float pb = pos[(di * 31 + dj) * 4 + head];
            sS[lr * 256 + kt] = (__bf16)(acc[r] + pb);
          }
        }
      }
      __syncthreads();

      // Row softmax: 4 threads per row, 64 contiguous cols each, v8bf accesses.
      {
        const int row = t >> 2;
        const int part = t & 3;
        __bf16* seg = sS + row * 256 + part * 64;
        float mx = -3.0e38f;
#pragma unroll
        for (int c = 0; c < 64; c += 8) {
          v8bf v = *(const v8bf*)(seg + c);
#pragma unroll
          for (int u = 0; u < 8; ++u) mx = fmaxf(mx, (float)v[u]);
        }
        sRed[t] = mx;
        __syncthreads();
        float rmx = fmaxf(fmaxf(sRed[row * 4 + 0], sRed[row * 4 + 1]),
                          fmaxf(sRed[row * 4 + 2], sRed[row * 4 + 3]));
        float sum = 0.f;
#pragma unroll
        for (int c = 0; c < 64; c += 8) {
          v8bf v = *(const v8bf*)(seg + c);
          v8bf o;
#pragma unroll
          for (int u = 0; u < 8; ++u) {
            float e = __expf((float)v[u] - rmx);
            sum += e;
            o[u] = (__bf16)e;
          }
          *(v8bf*)(seg + c) = o;
        }
        __syncthreads();  // all max reads done before sRed reuse
        sRed[t] = sum;
        __syncthreads();
        float rs = (sRed[row * 4 + 0] + sRed[row * 4 + 1]) +
                   (sRed[row * 4 + 2] + sRed[row * 4 + 3]);
        float is = 1.f / rs;
#pragma unroll
        for (int c = 0; c < 64; c += 8) {
          v8bf v = *(const v8bf*)(seg + c);
          v8bf o;
#pragma unroll
          for (int u = 0; u < 8; ++u) o[u] = (__bf16)((float)v[u] * is);
          *(v8bf*)(seg + c) = o;
        }
      }
      __syncthreads();

      // O = P V : one 16x16 tile per wave (4 M x 2 N), K=256 in 8 chunks.
      {
        const int tm = wave >> 1;
        const int tn = wave & 1;
        v8f acc = {};
#pragma unroll
        for (int kc = 0; kc < 8; ++kc) {
          v16bf a = load_a_frag(sS, 256, tm * 16, kc * 32, lane);
          v16bf bb = load_b_frag(sVTh, 256, tn * 16, kc * 32, lane);
          acc = WMMA_BF16(a, bb, acc);
        }
#pragma unroll
        for (int r = 0; r < 8; ++r) {
          int lr = tm * 16 + r + ((lane >> 4) << 3);
          int tok = row0 + lr;
          int ch = head * 32 + tn * 16 + (lane & 15);
          int ii = tok >> 4, jj = tok & 15;
          size_t g =
              ((size_t)b * 65536 + (size_t)(wi * 16 + ii) * 256 + (wj * 16 + jj)) * 128 + ch;
          out[g] = x[g] + acc[r];
        }
      }
      __syncthreads();
    }
  }
}

// ---------------------------------------------------------------------------
extern "C" void kernel_launch(void* const* d_in, const int* in_sizes, int n_in,
                              void* d_out, int out_size, void* d_ws, size_t ws_size,
                              hipStream_t stream) {
  const float* x     = (const float*)d_in[0];
  const float* n1g   = (const float*)d_in[1];
  const float* n1b   = (const float*)d_in[2];
  const float* qkv_w = (const float*)d_in[3];
  const float* qkv_b = (const float*)d_in[4];
  const float* pp_w  = (const float*)d_in[5];
  const float* pp_b  = (const float*)d_in[6];
  const float* l1_g  = (const float*)d_in[7];
  const float* l1_b  = (const float*)d_in[8];
  const float* f1_w  = (const float*)d_in[9];
  const float* f1_b  = (const float*)d_in[10];
  const float* l2_g  = (const float*)d_in[11];
  const float* l2_b  = (const float*)d_in[12];
  const float* f2_w  = (const float*)d_in[13];
  const float* f2_b  = (const float*)d_in[14];
  const float* l3_g  = (const float*)d_in[15];
  const float* l3_b  = (const float*)d_in[16];
  const float* f3_w  = (const float*)d_in[17];
  const float* f3_b  = (const float*)d_in[18];
  float* out = (float*)d_out;

  const int NW = 3 * 128 * 128;  // qkv weight elements
  __bf16* wqkv_bf = (__bf16*)d_ws;
  float* pos = (float*)((char*)d_ws + (size_t)NW * 2);

  conv_bf16_kernel<<<(NW + 255) / 256, 256, 0, stream>>>(qkv_w, wqkv_bf, NW);
  dpb_kernel<<<4, 256, 0, stream>>>(pp_w, pp_b, l1_g, l1_b, f1_w, f1_b,
                                    l2_g, l2_b, f2_w, f2_b, l3_g, l3_b,
                                    f3_w, f3_b, pos);

  int B = in_sizes[0] / (256 * 256 * 128);  // batch from x's flat size
  if (B < 1) B = 1;
  cf_attn_kernel<<<B * 256, 256, 0, stream>>>(x, n1g, n1b, wqkv_bf, qkv_b, pos, out);
}
